// TemporalAttentionModel_49417893708421
// MI455X (gfx1250) — compile-verified, matrix operand order checked by reading, not run on hardware
//
#include <hip/hip_runtime.h>
#include <hip/hip_bf16.h>

// ---------------------------------------------------------------------------
// TemporalAttentionModel for MI455X (gfx1250): all GEMMs on
// v_wmma_f32_16x16x32_bf16 (wave32 WMMA), f32 accum, f32 softmax/GELU.
// TDM (tensor_load_to_lds) used for the FFN O-tile staging.
// ---------------------------------------------------------------------------

#define Bdim 8
#define Ddim 256
#define Ndim 512
#define Tdim 64
#define Hdim 8
#define HDdim 32

typedef __attribute__((ext_vector_type(16))) __bf16 v16bf;
typedef __attribute__((ext_vector_type(8)))  float  v8f;
typedef __attribute__((ext_vector_type(4)))  unsigned int v4u;
typedef __attribute__((ext_vector_type(8)))  int    v8i;
typedef __attribute__((ext_vector_type(4)))  int    v4i;

#if defined(__gfx1250__) && __has_builtin(__builtin_amdgcn_tensor_load_to_lds)
#define USE_TDM 1
#else
#define USE_TDM 0
#endif

union FragU { v16bf v; uint4 q[2]; };
union Pack8 { unsigned short s[8]; uint4 q; };

// float -> bf16 (round to nearest even), as raw u16
__device__ __forceinline__ unsigned short f2bf(float f) {
    unsigned int u = __float_as_uint(f);
    u += 0x7FFFu + ((u >> 16) & 1u);
    return (unsigned short)(u >> 16);
}

__device__ __forceinline__ float gelu_exact(float x) {
    return 0.5f * x * (1.0f + erff(x * 0.7071067811865476f));
}

// Load one 16x32 bf16 fragment (A layout; also B when fed a transposed matrix):
// lane r (r=lane&15) = row; K split per ISA: kh = (lane>>4)*8,
// elems 0..7 = K[kh..kh+7], elems 8..15 = K[16+kh .. 16+kh+7].
__device__ __forceinline__ v16bf load_frag(const unsigned short* p0, int ld, int lane) {
    const int r  = lane & 15;
    const int kh = (lane >> 4) << 3;
    const unsigned short* p = p0 + r * ld + kh;
    FragU f;
    f.q[0] = *(const uint4*)(p);
    f.q[1] = *(const uint4*)(p + 16);
    return f.v;
}

__device__ __forceinline__ v8f wmma_bf16(v16bf a, v16bf b, v8f c) {
    return __builtin_amdgcn_wmma_f32_16x16x32_bf16(
        /*neg_a=*/false, a, /*neg_b=*/false, b,
        /*c_mod=*/(short)0, c, /*reuse_a=*/false, /*reuse_b=*/false);
}

// Workspace layout (in unsigned short elements)
#define WS_W12T 0
#define WS_W13T (WS_W12T + 256*256)
#define WS_W14T (WS_W13T + 256*256)
#define WS_W15T (WS_W14T + 256*256)          // [1024][256]
#define WS_W16T (WS_W15T + 1024*256)         // [256][1024]
#define WS_Q    (WS_W16T + 256*1024)         // [B][N][H][T][HD]
#define WS_K    (WS_Q + (size_t)Bdim*Ndim*Hdim*Tdim*HDdim)
#define WS_VT   (WS_K + (size_t)Bdim*Ndim*Hdim*Tdim*HDdim)  // [B][N][H][HD][T]
#define WS_O    (WS_VT + (size_t)Bdim*Ndim*Hdim*Tdim*HDdim) // [B][N][T][D]

// ---------------------------------------------------------------------------
// Kernel 1: transpose+convert weights to bf16 [out][in]
// ---------------------------------------------------------------------------
__global__ __launch_bounds__(256) void wprep_kernel(
    const float* __restrict__ W12, const float* __restrict__ W13,
    const float* __restrict__ W14, const float* __restrict__ W15,
    const float* __restrict__ W16, unsigned short* __restrict__ ws) {
    int i = blockIdx.x * blockDim.x + threadIdx.x;
    if (i < 3 * 65536) {
        int p = i >> 16, r = i & 65535;
        int o = r >> 8, in = r & 255;                 // Wt[o][in]
        const float* W = (p == 0) ? W12 : (p == 1) ? W13 : W14;
        ws[WS_W12T + (size_t)p * 65536 + r] = f2bf(W[in * 256 + o]);
    } else if (i < 3 * 65536 + 262144) {
        int j = i - 3 * 65536;                         // W15t [1024][256]
        int o = j >> 8, in = j & 255;
        ws[WS_W15T + j] = f2bf(W15[in * 1024 + o]);
    } else if (i < 3 * 65536 + 2 * 262144) {
        int j = i - (3 * 65536 + 262144);              // W16t [256][1024]
        int o = j >> 10, in = j & 1023;
        ws[WS_W16T + j] = f2bf(W16[in * 256 + o]);
    }
}

// ---------------------------------------------------------------------------
// Kernel 2: fused transpose-in + QKV projections + bias + GELU.
// One block per (b,n); 256 threads = 8 waves. 1536 WMMAs per block.
// ---------------------------------------------------------------------------
#define XLD 264   // token tile row stride (bf16), keeps 16B alignment (264*2=528)
__global__ __launch_bounds__(256) void qkv_kernel(
    const float* __restrict__ X, const unsigned short* __restrict__ ws,
    const float* __restrict__ b12, const float* __restrict__ b13,
    const float* __restrict__ b14,
    unsigned short* __restrict__ q, unsigned short* __restrict__ k,
    unsigned short* __restrict__ vt) {
    __shared__ __align__(16) unsigned short x_lds[Tdim * XLD];   // 33 KB

    const int bn = blockIdx.x;
    const int n = bn & (Ndim - 1);
    const int b = bn >> 9;

    // Load X[b, d, n, 0..63] (contiguous in t) and transpose into x_lds[t][d]
    {
        const int d = threadIdx.x;
        const float4* xp = (const float4*)(X + ((((size_t)b * Ddim + d) * Ndim + n) * Tdim));
        #pragma unroll
        for (int c = 0; c < 16; ++c) {
            float4 f = xp[c];
            int t = c * 4;
            x_lds[(t + 0) * XLD + d] = f2bf(f.x);
            x_lds[(t + 1) * XLD + d] = f2bf(f.y);
            x_lds[(t + 2) * XLD + d] = f2bf(f.z);
            x_lds[(t + 3) * XLD + d] = f2bf(f.w);
        }
    }
    __syncthreads();

    const int wv = threadIdx.x >> 5;
    const int lane = threadIdx.x & 31;
    const int nl = lane & 15;
    const int hi = (lane >> 4) << 3;
    const size_t headstride = (size_t)Tdim * HDdim;   // 2048

    // 3 projections x (4 mtiles x 16 ntiles) = 192 tile jobs over 8 waves
    for (int job = wv; job < 192; job += 8) {
        const int p  = job >> 6;
        const int rem = job & 63;
        const int mt = rem >> 4;
        const int nt = rem & 15;
        const unsigned short* Wt = ws + WS_W12T + (size_t)p * 65536;
        const float* bias = (p == 0) ? b12 : (p == 1) ? b13 : b14;
        const float bv = bias[nt * 16 + nl];
        v8f acc = {bv, bv, bv, bv, bv, bv, bv, bv};

        __builtin_prefetch(Wt + (size_t)nt * 16 * 256, 0, 1);   // global_prefetch_b8
        #pragma unroll
        for (int k0 = 0; k0 < 256; k0 += 32) {
            v16bf a  = load_frag(x_lds + mt * 16 * XLD + k0, XLD, lane);
            v16bf bf = load_frag(Wt + (size_t)nt * 16 * 256 + k0, 256, lane);
            acc = wmma_bf16(a, bf, acc);
        }
        const int d = nt * 16 + nl;
        const int h = d >> 5, hd = d & 31;
        const size_t hb = ((((size_t)b * Ndim + n) * Hdim + h)) * headstride;
        if (p == 2) {
            // V transposed: t runs along j -> 8 contiguous bf16 -> one b128 store
            Pack8 pk;
            #pragma unroll
            for (int j = 0; j < 8; ++j) pk.s[j] = f2bf(gelu_exact(acc[j]));
            *(uint4*)(vt + hb + (size_t)hd * Tdim + mt * 16 + hi) = pk.q;
        } else {
            unsigned short* dst = (p == 0) ? q : k;
            #pragma unroll
            for (int j = 0; j < 8; ++j) {
                const int t = mt * 16 + j + hi;
                dst[hb + (size_t)t * HDdim + hd] = f2bf(gelu_exact(acc[j]));
            }
        }
    }
}

// ---------------------------------------------------------------------------
// Kernel 3: causal attention per (b,n,h). 128 threads = 4 waves.
// scores = QK^T (WMMA), f32 softmax in LDS, out = P V (WMMA).
// ---------------------------------------------------------------------------
__global__ __launch_bounds__(128) void attn_kernel(
    const unsigned short* __restrict__ q, const unsigned short* __restrict__ k,
    const unsigned short* __restrict__ vt, unsigned short* __restrict__ o) {
    __shared__ __align__(16) float          s_sc[Tdim * Tdim];          // 16 KB
    __shared__ __align__(16) unsigned short s_pr[Tdim * Tdim];          // 8 KB

    const int idx = blockIdx.x;
    const int h = idx & 7;
    const int n = (idx >> 3) & (Ndim - 1);
    const int b = idx >> 12;
    const size_t hb = ((((size_t)b * Ndim + n) * Hdim + h)) * (size_t)(Tdim * HDdim);
    const unsigned short* qh = q + hb;
    const unsigned short* kh = k + hb;
    const unsigned short* vh = vt + hb;

    const int wv = threadIdx.x >> 5;
    const int lane = threadIdx.x & 31;
    const int nl = lane & 15;
    const int hi = (lane >> 4) << 3;

    // --- scores = Q K^T * 1/sqrt(HD), causal mask ---
    v16bf afr = load_frag(qh + wv * 16 * HDdim, HDdim, lane);
    #pragma unroll
    for (int ntile = 0; ntile < 4; ++ntile) {
        v16bf bfr = load_frag(kh + ntile * 16 * HDdim, HDdim, lane);
        v8f acc = {};
        acc = wmma_bf16(afr, bfr, acc);
        #pragma unroll
        for (int j = 0; j < 8; ++j) {
            const int t = wv * 16 + j + hi;
            const int s = ntile * 16 + nl;
            const float val = acc[j] * 0.17677669529663687f;   // 1/sqrt(32)
            s_sc[t * Tdim + s] = (s <= t) ? val : -9.0e15f;
        }
    }
    __syncthreads();

    // --- softmax over each row (64 rows, one thread per row) ---
    if (threadIdx.x < Tdim) {
        const int r = threadIdx.x;
        float mx = -3.4e38f;
        for (int s = 0; s < Tdim; ++s) mx = fmaxf(mx, s_sc[r * Tdim + s]);
        float sum = 0.0f;
        for (int s = 0; s < Tdim; ++s) sum += __expf(s_sc[r * Tdim + s] - mx);
        const float inv = 1.0f / sum;
        for (int s = 0; s < Tdim; ++s)
            s_pr[r * Tdim + s] = f2bf(__expf(s_sc[r * Tdim + s] - mx) * inv);
    }
    __syncthreads();

    // --- out = P V  (P: 64x64 bf16 in LDS; V^T rows contiguous) ---
    #pragma unroll
    for (int ntile = 0; ntile < 2; ++ntile) {
        v8f acc = {};
        #pragma unroll
        for (int s0 = 0; s0 < Tdim; s0 += 32) {
            v16bf ap = load_frag(s_pr + wv * 16 * Tdim + s0, Tdim, lane);
            v16bf bv = load_frag(vh + ntile * 16 * Tdim + s0, Tdim, lane);
            acc = wmma_bf16(ap, bv, acc);
        }
        #pragma unroll
        for (int j = 0; j < 8; ++j) {
            const int t = wv * 16 + j + hi;
            const int d = h * HDdim + ntile * 16 + nl;
            o[((((size_t)b * Ndim + n) * Tdim + t)) * Ddim + d] = f2bf(acc[j]);
        }
    }
}

// ---------------------------------------------------------------------------
// Kernel 4: FFN  out = gelu(O @ W15 + b15) @ W16 + b16, transposed store.
// One block per (b,n); 8 waves; C tiles live in registers across 4 K-chunks.
// O-tile staged into LDS via the Tensor Data Mover when available.
// ---------------------------------------------------------------------------
__global__ __launch_bounds__(256) void ffn_kernel(
    const unsigned short* __restrict__ o, const unsigned short* __restrict__ w15t,
    const unsigned short* __restrict__ w16t, const float* __restrict__ b15,
    const float* __restrict__ b16, float* __restrict__ out) {
    __shared__ __align__(16) unsigned short lds_o[Tdim * Ddim];  // 32 KB
    __shared__ __align__(16) unsigned short lds_h[Tdim * Ddim];  // 32 KB

    const int bn = blockIdx.x;
    const int n = bn & (Ndim - 1);
    const int b = bn >> 9;

#if USE_TDM
    // One TDM descriptor: 1 row x 8192 dwords (32 KB) global -> LDS.
    if (threadIdx.x < 32) {
        unsigned lds_base =
            (unsigned)(unsigned long long)(__attribute__((address_space(3))) unsigned short*)lds_o;
        unsigned long long g =
            (unsigned long long)(uintptr_t)(o + (size_t)bn * Tdim * Ddim);
        v4u g0;
        g0[0] = 1u;                                        // count=1 (valid user D#)
        g0[1] = lds_base;                                  // lds_addr (bytes)
        g0[2] = (unsigned)(g & 0xFFFFFFFFu);               // global_addr[31:0]
        g0[3] = (unsigned)((g >> 32) & 0x01FFFFFFu)        // global_addr[56:32]
                | 0x80000000u;                             // type=2 ("image")
        v8i g1;
        g1[0] = (int)(2u << 16);                           // data_size=4B; no multicast
        g1[1] = (int)(8192u << 16);                        // tensor_dim0[15:0]=8192
        g1[2] = (int)(1u << 16);                           // tensor_dim0 hi=0; tensor_dim1=1
        g1[3] = (int)(8192u << 16);                        // tile_dim0=8192
        g1[4] = 1;                                         // tile_dim1=1, tile_dim2=0
        g1[5] = 8192;                                      // tensor_dim0_stride (dwords)
        g1[6] = 0;
        g1[7] = 0;
        v4i gz4 = {0, 0, 0, 0};
        v8i gz8 = {0, 0, 0, 0, 0, 0, 0, 0};
        __builtin_amdgcn_tensor_load_to_lds(g0, g1, gz4, gz4, gz8, 0);
        __builtin_amdgcn_s_wait_tensorcnt(0);
    }
#else
    {
        const uint4* src = (const uint4*)(o + (size_t)bn * Tdim * Ddim);
        uint4* dst = (uint4*)lds_o;
        for (int i = threadIdx.x; i < (Tdim * Ddim) / 8; i += 256) dst[i] = src[i];
    }
#endif
    __syncthreads();

    const int wv = threadIdx.x >> 5;
    const int lane = threadIdx.x & 31;
    const int nl = lane & 15;
    const int hi = (lane >> 4) << 3;

    v8f acc[2][4];

    for (int c = 0; c < 4; ++c) {
        // H1 chunk: gelu(O @ W15[:, c*256 .. c*256+255] + b15)
        #pragma unroll
        for (int i = 0; i < 8; ++i) {
            const int job = wv * 8 + i;       // 0..63
            const int mt = job >> 4;
            const int nt = job & 15;
            const int ocol = c * 256 + nt * 16 + nl;
            const float bv = b15[ocol];
            v8f hacc = {bv, bv, bv, bv, bv, bv, bv, bv};
            const unsigned short* wrow = w15t + (size_t)(c * 256 + nt * 16) * 256;
            __builtin_prefetch(wrow, 0, 1);
            #pragma unroll
            for (int k0 = 0; k0 < 256; k0 += 32) {
                v16bf a  = load_frag(lds_o + mt * 16 * Ddim + k0, Ddim, lane);
                v16bf bf = load_frag(wrow + k0, 256, lane);
                hacc = wmma_bf16(a, bf, hacc);
            }
            #pragma unroll
            for (int j = 0; j < 8; ++j) {
                const int m = mt * 16 + j + hi;
                lds_h[m * Ddim + nt * 16 + nl] = f2bf(gelu_exact(hacc[j]));
            }
        }
        __syncthreads();

        // accumulate: C += H1_chunk @ W16[c*256 .. , :]
        #pragma unroll
        for (int u = 0; u < 2; ++u) {
            const int nto = wv * 2 + u;       // 0..15
            #pragma unroll
            for (int mt = 0; mt < 4; ++mt) {
                v8f a2;
                if (c == 0) {
                    const float bv = b16[nto * 16 + nl];
                    a2 = (v8f){bv, bv, bv, bv, bv, bv, bv, bv};
                } else {
                    a2 = acc[u][mt];
                }
                const unsigned short* wrow = w16t + (size_t)(nto * 16) * 1024 + c * 256;
                #pragma unroll
                for (int k0 = 0; k0 < 256; k0 += 32) {
                    v16bf a  = load_frag(lds_h + mt * 16 * Ddim + k0, Ddim, lane);
                    v16bf bf = load_frag(wrow + k0, 1024, lane);
                    a2 = wmma_bf16(a, bf, a2);
                }
                acc[u][mt] = a2;
            }
        }
        __syncthreads();
    }

    // store transposed to (B, D, N, T) f32; t runs along j -> contiguous 8 floats
    #pragma unroll
    for (int u = 0; u < 2; ++u) {
        const int nto = wv * 2 + u;
        #pragma unroll
        for (int mt = 0; mt < 4; ++mt) {
            const int d = nto * 16 + nl;
            float* dst = out + (((size_t)b * Ddim + d) * Ndim + n) * Tdim + mt * 16 + hi;
            float4 lo = make_float4(acc[u][mt][0], acc[u][mt][1], acc[u][mt][2], acc[u][mt][3]);
            float4 hi4 = make_float4(acc[u][mt][4], acc[u][mt][5], acc[u][mt][6], acc[u][mt][7]);
            *(float4*)(dst)     = lo;
            *(float4*)(dst + 4) = hi4;
        }
    }
}

// ---------------------------------------------------------------------------
extern "C" void kernel_launch(void* const* d_in, const int* in_sizes, int n_in,
                              void* d_out, int out_size, void* d_ws, size_t ws_size,
                              hipStream_t stream) {
    const float* X   = (const float*)d_in[0];
    const float* W12 = (const float*)d_in[1];
    const float* b12 = (const float*)d_in[2];
    const float* W13 = (const float*)d_in[3];
    const float* b13 = (const float*)d_in[4];
    const float* W14 = (const float*)d_in[5];
    const float* b14 = (const float*)d_in[6];
    const float* W15 = (const float*)d_in[7];
    const float* b15 = (const float*)d_in[8];
    const float* W16 = (const float*)d_in[9];
    const float* b16 = (const float*)d_in[10];

    unsigned short* ws = (unsigned short*)d_ws;
    unsigned short* qbuf  = ws + WS_Q;
    unsigned short* kbuf  = ws + WS_K;
    unsigned short* vtbuf = ws + WS_VT;
    unsigned short* obuf  = ws + WS_O;

    // 1) weights -> bf16 transposed
    wprep_kernel<<<2816, 256, 0, stream>>>(W12, W13, W14, W15, W16, ws);
    // 2) QKV projections (+transpose-in, bias, GELU)
    qkv_kernel<<<Bdim * Ndim, 256, 0, stream>>>(X, ws, b12, b13, b14, qbuf, kbuf, vtbuf);
    // 3) causal attention per head
    attn_kernel<<<Bdim * Ndim * Hdim, 128, 0, stream>>>(qbuf, kbuf, vtbuf, obuf);
    // 4) FFN + transposed f32 store
    ffn_kernel<<<Bdim * Ndim, 256, 0, stream>>>(obuf, ws + WS_W15T, ws + WS_W16T,
                                                b15, b16, (float*)d_out);
}
